// PointNet2BackboneLight_35802847379563
// MI455X (gfx1250) — matrix operand order repaired and checked by reading, not compiled, not run on hardware
//
#include <hip/hip_runtime.h>
#include <hip/hip_bf16.h>

typedef __attribute__((ext_vector_type(16))) _Float16 v16h;
typedef __attribute__((ext_vector_type(8)))  _Float16 h8;
typedef __attribute__((ext_vector_type(8)))  float    v8f;

// ======================================================================
// Farthest point sampling: one block per batch. Point cloud + d2 in LDS.
// Matches jax scan semantics: inds[0]=0; inds[t]=argmax(d2 after update
// with inds[t-1]); argmax tie-break = lowest index (jnp.argmax).
// ======================================================================
__global__ __launch_bounds__(256) void fps_kernel(const float* __restrict__ xyz,
                                                  int N, int npoint,
                                                  int* __restrict__ inds,
                                                  float* __restrict__ new_xyz) {
  __shared__ float sx[2048 * 3];
  __shared__ float sd2[2048];
  __shared__ float swv[8];
  __shared__ int   swi[8];
  const int b = blockIdx.x, tid = threadIdx.x, lane = tid & 31, wv = tid >> 5;
  const float* px = xyz + (size_t)b * N * 3;
  for (int i = tid; i < N * 3; i += 256) sx[i] = px[i];
  for (int i = tid; i < N; i += 256) sd2[i] = 1e10f;
  __syncthreads();
  int last = 0;
  int* pi = inds + (size_t)b * npoint;
  float* pnx = new_xyz + (size_t)b * npoint * 3;
  for (int t = 0; t < npoint; ++t) {
    float lx = sx[last * 3 + 0], ly = sx[last * 3 + 1], lz = sx[last * 3 + 2];
    if (tid == 0) { pi[t] = last; pnx[t*3+0] = lx; pnx[t*3+1] = ly; pnx[t*3+2] = lz; }
    float bv = -1.f; int bi = 0;
    for (int i = tid; i < N; i += 256) {
      float dx = sx[i*3+0] - lx, dy = sx[i*3+1] - ly, dz = sx[i*3+2] - lz;
      float nd = fminf(sd2[i], dx*dx + dy*dy + dz*dz);
      sd2[i] = nd;
      if (nd > bv || (nd == bv && i < bi)) { bv = nd; bi = i; }
    }
    for (int off = 16; off > 0; off >>= 1) {
      float ov = __shfl_down(bv, off);
      int   oi = __shfl_down(bi, off);
      if (ov > bv || (ov == bv && oi < bi)) { bv = ov; bi = oi; }
    }
    if (lane == 0) { swv[wv] = bv; swi[wv] = bi; }
    __syncthreads();
    bv = swv[0]; bi = swi[0];
#pragma unroll
    for (int w = 1; w < 8; ++w) {
      float ov = swv[w]; int oi = swi[w];
      if (ov > bv || (ov == bv && oi < bi)) { bv = ov; bi = oi; }
    }
    last = bi;
    __syncthreads();
  }
}

// ======================================================================
// Ball query (CUDA semantics): first K=32 indices (in order) with d2<r2,
// tail filled with first hit (0 if none). One wave32 per center.
// ======================================================================
__global__ __launch_bounds__(256) void ballquery_kernel(const float* __restrict__ xyz,
                                                        const float* __restrict__ new_xyz,
                                                        int N, int S, int total, float r2,
                                                        int* __restrict__ gidx) {
  const int K = 32;
  int gid = blockIdx.x * 8 + (threadIdx.x >> 5);
  if (gid >= total) return;
  const int lane = threadIdx.x & 31;
  const int b = gid / S;
  const float* px = xyz + (size_t)b * N * 3;
  float cx = new_xyz[(size_t)gid*3+0], cy = new_xyz[(size_t)gid*3+1], cz = new_xyz[(size_t)gid*3+2];
  int* row = gidx + (size_t)gid * K;
  int tot = 0, first = 0;
  bool have = false;
  for (int base = 0; base < N; base += 32) {
    int i = base + lane;
    float dx = px[i*3+0] - cx, dy = px[i*3+1] - cy, dz = px[i*3+2] - cz;
    bool pred = (dx*dx + dy*dy + dz*dz) < r2;
    unsigned mask = (unsigned)__ballot(pred ? 1 : 0);
    if (!have && mask) { int fl = __ffs(mask) - 1; first = __shfl(i, fl); have = true; }
    int slot = tot + __popc(mask & ((1u << lane) - 1u));
    if (pred && slot < K) row[slot] = i;
    tot += __popc(mask);
    if (tot >= K) break;
  }
  if (tot < K)
    for (int j = tot + lane; j < K; j += 32) row[j] = first;
}

// ======================================================================
// One WMMA MLP layer over 64 points resident in LDS.
// A (weights, row-major [COUT][CIN] f16): two h8
//   loads per lane (K 0-15 for lanes 0-15, 16-31 for 16-31).
// B (activations): lane = column (point), 16 contiguous K halves.
// D: vgpr v -> out channel mt*16 + v + 8*(lane>>4), point = ntile*16+(lane&15).
// ======================================================================
template<int CIN, int COUT, int SIN, int SOUT>
static __device__ __forceinline__ void mlp_layer(const _Float16* __restrict__ Xin,
                                                 _Float16* __restrict__ Yout,
                                                 const _Float16* __restrict__ W,
                                                 const float* __restrict__ bias, int tid) {
  const int wave = tid >> 5, lane = tid & 31;
  const int ntile = wave & 3, mstart = wave >> 2;
  const int col = lane & 15, khalf = lane >> 4;
  const int MT = COUT / 16, KT = CIN / 32;
  const int p = ntile * 16 + col;
  const _Float16* wrow = W + (size_t)(0 * 16 + col) * CIN;  // row set per mt below
  const _Float16* xrow = Xin + p * SIN + khalf * 16;
  for (int mt = mstart; mt < MT; mt += 2) {
    const _Float16* wr = W + (size_t)(mt * 16 + col) * CIN;
    v8f acc = {};
#pragma unroll
    for (int kt = 0; kt < KT; ++kt) {
      const int k0 = kt * 32;
      h8 alo = *(const h8*)(wr + k0 + khalf * 16);
      h8 ahi = *(const h8*)(wr + k0 + 8 + khalf * 16);
      v16h a;
#pragma unroll
      for (int e = 0; e < 8; ++e) { a[e] = alo[e]; a[e + 8] = ahi[e]; }
      v16h bb = *(const v16h*)(xrow + k0);
      acc = __builtin_amdgcn_wmma_f32_16x16x32_f16(false, a, false, bb, (short)0, acc,
                                                   false, false);
    }
    const int obase = mt * 16 + khalf * 8;
    h8 o;
#pragma unroll
    for (int v = 0; v < 8; ++v) {
      float x = acc[v] + bias[obase + v];
      o[v] = (_Float16)fmaxf(x, 0.f);
    }
    *(h8*)(Yout + p * SOUT + obase) = o;
  }
  (void)wrow;
}

// ======================================================================
// Fused: gather groups -> 3-layer MLP (WMMA, f16, bias+ReLU) -> maxpool.
// 2 groups (64 points) per 256-thread block. All activations in LDS.
// ======================================================================
template<int CFEAT, int CIN_PAD, int D1, int D2, int D3>
__global__ __launch_bounds__(256) void sa_mlp_kernel(
    const float* __restrict__ xyz, const _Float16* __restrict__ feats,
    const float* __restrict__ new_xyz, const int* __restrict__ gidx,
    int N, float inv_radius,
    const _Float16* __restrict__ w1, const float* __restrict__ b1,
    const _Float16* __restrict__ w2, const float* __restrict__ b2,
    const _Float16* __restrict__ w3, const float* __restrict__ b3,
    _Float16* __restrict__ out_feats, int S) {
  constexpr int ST0 = (CIN_PAD > D2 ? CIN_PAD : D2) + 16;  // buf0 row stride (halves)
  constexpr int ST1 = (D1 > D3 ? D1 : D3) + 16;            // buf1 row stride
  extern __shared__ __align__(128) char smem_raw[];
  _Float16* buf0 = (_Float16*)smem_raw;
  _Float16* buf1 = buf0 + 64 * ST0;
  __shared__ int   sIdx[64];
  __shared__ float sC[2][3];
  const int tid = threadIdx.x;
  const int sBase = blockIdx.x * 2;  // flat index over B*S (both groups same batch)
  const int b = sBase / S;
  if (tid < 64) sIdx[tid] = gidx[(size_t)(sBase + (tid >> 5)) * 32 + (tid & 31)];
  if (tid < 6)  sC[tid / 3][tid % 3] = new_xyz[(size_t)sBase * 3 + tid];
  __syncthreads();
  // Stage 0: gather [64][CIN_PAD] into LDS (rel-xyz/r, feats, zero pad).
  for (int i = tid; i < 64 * CIN_PAD; i += 256) {
    int p = i / CIN_PAD, c = i % CIN_PAD;
    int idx = sIdx[p];
    float v = 0.f;
    if (c < 3)
      v = (xyz[((size_t)b * N + idx) * 3 + c] - sC[p >> 5][c]) * inv_radius;
    else if (CFEAT > 0 && c < 3 + CFEAT)
      v = (float)feats[((size_t)b * N + idx) * CFEAT + (c - 3)];
    buf0[p * ST0 + c] = (_Float16)v;
  }
  __syncthreads();
  mlp_layer<CIN_PAD, D1, ST0, ST1>(buf0, buf1, w1, b1, tid);
  __syncthreads();
  mlp_layer<D1, D2, ST1, ST0>(buf1, buf0, w2, b2, tid);
  __syncthreads();
  mlp_layer<D2, D3, ST0, ST1>(buf0, buf1, w3, b3, tid);
  __syncthreads();
  // Maxpool each group of 32 samples -> out_feats [B*S][D3] (point-major f16).
  for (int i = tid; i < 2 * D3; i += 256) {
    int g = i / D3, o = i % D3;
    float m = -3.4e38f;
#pragma unroll 4
    for (int p = g * 32; p < g * 32 + 32; ++p) m = fmaxf(m, (float)buf1[p * ST1 + o]);
    out_feats[((size_t)(sBase + g)) * D3 + o] = (_Float16)m;
  }
}

// ======================================================================
// Final 256x256 projection (WMMA, +bias, no ReLU), f32 output [B,256,S].
// One block = 64 points.
// ======================================================================
__global__ __launch_bounds__(256) void cov_kernel(const _Float16* __restrict__ f,
                                                  const _Float16* __restrict__ W,
                                                  const float* __restrict__ bias,
                                                  int S, float* __restrict__ out) {
  constexpr int C = 256, ST = C + 16;
  extern __shared__ __align__(128) char smem_raw[];
  _Float16* X = (_Float16*)smem_raw;  // 64*ST halves
  const int tid = threadIdx.x;
  const int pBase = blockIdx.x * 64;  // flat over B*S
  const int b = pBase / S, sloc = pBase % S;
  for (int i = tid; i < 64 * C; i += 256) {
    int p = i >> 8, c = i & 255;
    X[p * ST + c] = f[(size_t)(pBase + p) * C + c];
  }
  __syncthreads();
  const int wave = tid >> 5, lane = tid & 31;
  const int ntile = wave & 3, mstart = wave >> 2;
  const int col = lane & 15, khalf = lane >> 4;
  const int p = ntile * 16 + col;
  const _Float16* xrow = X + p * ST + khalf * 16;
  for (int mt = mstart; mt < 16; mt += 2) {
    const _Float16* wr = W + (size_t)(mt * 16 + col) * C;
    v8f acc = {};
#pragma unroll
    for (int kt = 0; kt < 8; ++kt) {
      const int k0 = kt * 32;
      h8 alo = *(const h8*)(wr + k0 + khalf * 16);
      h8 ahi = *(const h8*)(wr + k0 + 8 + khalf * 16);
      v16h a;
#pragma unroll
      for (int e = 0; e < 8; ++e) { a[e] = alo[e]; a[e + 8] = ahi[e]; }
      v16h bb = *(const v16h*)(xrow + k0);
      acc = __builtin_amdgcn_wmma_f32_16x16x32_f16(false, a, false, bb, (short)0, acc,
                                                   false, false);
    }
    const int obase = mt * 16 + khalf * 8;
    const int sg = sloc + p;
#pragma unroll
    for (int v = 0; v < 8; ++v) {
      int o = obase + v;
      out[((size_t)b * C + o) * S + sg] = acc[v] + bias[o];
    }
  }
}

// ======================================================================
// Weight conversion: f32 [cout][cin] * scale[cout] -> f16 [cout][cin_pad].
// ======================================================================
__global__ void convert_w_kernel(const float* __restrict__ W, const float* __restrict__ s,
                                 int cout, int cin, int cinp, _Float16* __restrict__ out) {
  int i = blockIdx.x * 256 + threadIdx.x;
  if (i >= cout * cinp) return;
  int o = i / cinp, c = i % cinp;
  float v = 0.f;
  if (c < cin) v = W[(size_t)o * cin + c] * (s ? s[o] : 1.f);
  out[i] = (_Float16)v;
}

// ======================================================================
// Seeds copy + index composition i0[i1[i2]].
// ======================================================================
__global__ void finalize_kernel(const float* __restrict__ xyz3, const int* __restrict__ i0,
                                const int* __restrict__ i1, const int* __restrict__ i2,
                                int S1, int S2, int S3, int total,
                                float* __restrict__ seeds, int* __restrict__ inds) {
  int i = blockIdx.x * 256 + threadIdx.x;
  if (i >= total) return;
  int b = i / S3;
  int a2 = i2[i];
  int a1 = i1[(size_t)b * S2 + a2];
  inds[i] = i0[(size_t)b * S1 + a1];
  seeds[i * 3 + 0] = xyz3[i * 3 + 0];
  seeds[i * 3 + 1] = xyz3[i * 3 + 1];
  seeds[i * 3 + 2] = xyz3[i * 3 + 2];
}

// ======================================================================
// Host orchestration
// ======================================================================
struct BranchWs {
  float *xyz1, *xyz2, *xyz3;
  int *i0, *i1, *i2, *g1, *g2, *g3;
  _Float16 *f1, *f2, *f3;
};

extern "C" void kernel_launch(void* const* d_in, const int* in_sizes, int n_in,
                              void* d_out, int out_size, void* d_ws, size_t ws_size,
                              hipStream_t stream) {
  (void)in_sizes; (void)n_in; (void)out_size; (void)ws_size;
  const int B = 32;
  char* ws = (char*)d_ws;
  size_t off = 0;
  auto alloc = [&](size_t bytes) -> char* {
    off = (off + 255) & ~(size_t)255;
    char* p = ws + off;
    off += bytes;
    return p;
  };

  // ---- workspace: converted weights (f16, padded) ----
  _Float16* w1a = (_Float16*)alloc((size_t)64  * 32  * 2);
  _Float16* w1b = (_Float16*)alloc((size_t)64  * 64  * 2);
  _Float16* w1c = (_Float16*)alloc((size_t)128 * 64  * 2);
  _Float16* w2a = (_Float16*)alloc((size_t)128 * 160 * 2);
  _Float16* w2b = (_Float16*)alloc((size_t)128 * 128 * 2);
  _Float16* w2c = (_Float16*)alloc((size_t)256 * 128 * 2);
  _Float16* w3a = (_Float16*)alloc((size_t)256 * 288 * 2);
  _Float16* w3b = (_Float16*)alloc((size_t)256 * 256 * 2);
  _Float16* w3c = (_Float16*)alloc((size_t)256 * 256 * 2);
  _Float16* wcv = (_Float16*)alloc((size_t)256 * 256 * 2);

  auto mkbranch = [&](int S1, int S2, int S3) -> BranchWs {
    BranchWs br;
    br.xyz1 = (float*)alloc((size_t)B * S1 * 3 * 4);
    br.xyz2 = (float*)alloc((size_t)B * S2 * 3 * 4);
    br.xyz3 = (float*)alloc((size_t)B * S3 * 3 * 4);
    br.i0 = (int*)alloc((size_t)B * S1 * 4);
    br.i1 = (int*)alloc((size_t)B * S2 * 4);
    br.i2 = (int*)alloc((size_t)B * S3 * 4);
    br.g1 = (int*)alloc((size_t)B * S1 * 32 * 4);
    br.g2 = (int*)alloc((size_t)B * S2 * 32 * 4);
    br.g3 = (int*)alloc((size_t)B * S3 * 32 * 4);
    br.f1 = (_Float16*)alloc((size_t)B * S1 * 128 * 2);
    br.f2 = (_Float16*)alloc((size_t)B * S2 * 256 * 2);
    br.f3 = (_Float16*)alloc((size_t)B * S3 * 256 * 2);
    return br;
  };
  BranchWs brS = mkbranch(1024, 512, 256);
  BranchWs brT = mkbranch(512, 256, 128);

  // ---- input pointers (setup_inputs order, tuples flattened) ----
  const float* searchPts = (const float*)d_in[0];
  const float* tmplPts   = (const float*)d_in[1];
  auto convW = [&](int wi, int si, int cout, int cin, int cinp, _Float16* dst) {
    int n = cout * cinp;
    convert_w_kernel<<<(n + 255) / 256, 256, 0, stream>>>(
        (const float*)d_in[wi], si >= 0 ? (const float*)d_in[si] : nullptr,
        cout, cin, cinp, dst);
  };
  // sa1: 2..10, sa2: 11..19, sa3: 20..28, cov: 29,30
  convW(2, 3, 64, 3, 32, w1a);    const float* b1a = (const float*)d_in[4];
  convW(5, 6, 64, 64, 64, w1b);   const float* b1b = (const float*)d_in[7];
  convW(8, 9, 128, 64, 64, w1c);  const float* b1c = (const float*)d_in[10];
  convW(11, 12, 128, 131, 160, w2a); const float* b2a = (const float*)d_in[13];
  convW(14, 15, 128, 128, 128, w2b); const float* b2b = (const float*)d_in[16];
  convW(17, 18, 256, 128, 128, w2c); const float* b2c = (const float*)d_in[19];
  convW(20, 21, 256, 259, 288, w3a); const float* b3a = (const float*)d_in[22];
  convW(23, 24, 256, 256, 256, w3b); const float* b3b = (const float*)d_in[25];
  convW(26, 27, 256, 256, 256, w3c); const float* b3c = (const float*)d_in[28];
  convW(29, -1, 256, 256, 256, wcv); const float* bcv = (const float*)d_in[30];

  // dynamic LDS per fused-MLP kernel: 64*(ST0+ST1)*2 bytes
  const size_t SM1 = (size_t)64 * (80 + 144) * 2;   // 28 KB
  const size_t SM2 = (size_t)64 * (176 + 272) * 2;  // 56 KB
  const size_t SM3 = (size_t)64 * (304 + 272) * 2;  // 72 KB
  const size_t SMC = (size_t)64 * 272 * 2;          // 34 KB
  const float R1 = 0.3f, R2 = 0.5f, R3 = 0.7f;

  auto run = [&](const float* pts, int N, int S1, int S2, int S3, BranchWs& br,
                 float* seeds, float* featsOut, int* indsOut) {
    fps_kernel<<<B, 256, 0, stream>>>(pts, N, S1, br.i0, br.xyz1);
    ballquery_kernel<<<(B * S1 + 7) / 8, 256, 0, stream>>>(pts, br.xyz1, N, S1, B * S1,
                                                           R1 * R1, br.g1);
    sa_mlp_kernel<0, 32, 64, 64, 128><<<B * S1 / 2, 256, SM1, stream>>>(
        pts, nullptr, br.xyz1, br.g1, N, 1.f / R1,
        w1a, b1a, w1b, b1b, w1c, b1c, br.f1, S1);

    fps_kernel<<<B, 256, 0, stream>>>(br.xyz1, S1, S2, br.i1, br.xyz2);
    ballquery_kernel<<<(B * S2 + 7) / 8, 256, 0, stream>>>(br.xyz1, br.xyz2, S1, S2,
                                                           B * S2, R2 * R2, br.g2);
    sa_mlp_kernel<128, 160, 128, 128, 256><<<B * S2 / 2, 256, SM2, stream>>>(
        br.xyz1, br.f1, br.xyz2, br.g2, S1, 1.f / R2,
        w2a, b2a, w2b, b2b, w2c, b2c, br.f2, S2);

    fps_kernel<<<B, 256, 0, stream>>>(br.xyz2, S2, S3, br.i2, br.xyz3);
    ballquery_kernel<<<(B * S3 + 7) / 8, 256, 0, stream>>>(br.xyz2, br.xyz3, S2, S3,
                                                           B * S3, R3 * R3, br.g3);
    sa_mlp_kernel<256, 288, 256, 256, 256><<<B * S3 / 2, 256, SM3, stream>>>(
        br.xyz2, br.f2, br.xyz3, br.g3, S2, 1.f / R3,
        w3a, b3a, w3b, b3b, w3c, b3c, br.f3, S3);

    cov_kernel<<<B * S3 / 64, 256, SMC, stream>>>(br.f3, wcv, bcv, S3, featsOut);
    finalize_kernel<<<(B * S3 + 255) / 256, 256, 0, stream>>>(
        br.xyz3, br.i0, br.i1, br.i2, S1, S2, S3, B * S3, seeds, indsOut);
  };

  // ---- d_out layout (flat concat, element offsets; float/int both 4B) ----
  float* fo = (float*)d_out;
  int*   io = (int*)d_out;
  // s_seeds 24576 | s_feats 2097152 | s_inds 8192 | t_seeds 12288 | t_feats 1048576 | t_inds 4096
  run(searchPts, 2048, 1024, 512, 256, brS, fo + 0, fo + 24576, io + 2121728);
  run(tmplPts, 1024, 512, 256, 128, brT, fo + 2129920, fo + 2142208, io + 3190784);
}
